// ExplicitMultiheadAttention_9766755631774
// MI455X (gfx1250) — matrix operand order repaired and checked by reading, not compiled, not run on hardware
//
#include <hip/hip_runtime.h>

// ExplicitMultiheadAttention for MI455X (gfx1250), wave32 + WMMA bf16 +
// async global->LDS copies (ASYNCcnt).
//
// Roofline: ~69 GFLOP vs ~50 MB unavoidable traffic -> compute bound; all
// matrix math goes through v_wmma_f32_16x16x32_bf16 (f32 accumulate, bf16
// keeps f32 exponent range). Attention is flash-style (scores never hit HBM).
//
// Data movement: tiles are copied global->LDS with
// GLOBAL_LOAD_ASYNC_TO_LDS_B128 (no VGPR staging, tracked by ASYNCcnt),
// double-buffered so the async engine fills buf^1 while WMMAs consume buf.
//
// Pipeline:
//   prep:  x -> bf16 (vectorized), W{q,k,v,o} -> bf16 transposed [out][in]
//   qkv :  128x64 double-buffered bf16 GEMM tiles, bias fused, writes
//          Q [B,H,L,d], K [B,H,L,d], V^T [B,H,d,L]
//   attn:  per (b,h,64q): stream 64-key blocks, online softmax, P.V accum
//   out :  128x64 bf16 GEMM, f32 + bias to d_out
// Workspace: xb 8MB | 4xWT 8MB | Q,K,Vt,ctx 32MB = 48MB.

#define D_MODEL 1024
#define NHEAD   16
#define D_HEAD  64
#define SEQ_L   2048
#define ATT_SCALE 0.125f   // 1/sqrt(64)

typedef __attribute__((ext_vector_type(16))) __bf16 v16bf;
typedef __attribute__((ext_vector_type(8)))  float  v8f;

union FragU { v16bf v; unsigned u[8]; };
union Pack8 { __bf16 h[8]; uint4 v; };

// ---- WMMA fragment loaders (layouts per CDNA5 ISA §7.12.2) -----
// A-frag (16x32 MxK): lane m = lane&15, half = lane>>4;
// VGPR v<4: K = 2v + 8*half, v>=4: K = 16 + 2(v-4) + 8*half.
__device__ __forceinline__ v16bf ld_a(const __bf16* p, int stride, int lane) {
  FragU f;
  const __bf16* row = p + (lane & 15) * stride + ((lane >> 4) << 3);
#pragma unroll
  for (int i = 0; i < 8; ++i) {
    int k = 2 * i + (i >= 4 ? 8 : 0);
    f.u[i] = *(const unsigned*)(row + k);
  }
  return f.v;
}
// B-frag (32x16 KxN) from a B^T tile [N][K]: lane n = lane&15, half = lane>>4;
// VGPR v<4: K = 2v + 8*half, v>=4: K = 16 + 2(v-4) + 8*half.
__device__ __forceinline__ v16bf ld_b(const __bf16* p, int stride, int lane) {
  FragU f;
  const __bf16* row = p + (lane & 15) * stride + ((lane >> 4) << 3);
#pragma unroll
  for (int i = 0; i < 8; ++i) f.u[i] = *(const unsigned*)(row + 2 * i + (i >= 4 ? 8 : 0));
  return f.v;
}
__device__ __forceinline__ v8f wmma_bf16(v16bf a, v16bf b, v8f c) {
  return __builtin_amdgcn_wmma_f32_16x16x32_bf16(false, a, false, b, (short)0, c,
                                                 false, false);
}

// ---- async global->LDS tile copies (CDNA5 GLOBAL_LOAD_ASYNC_TO_LDS) -------
// Per lane: copies 16B from the global address to the LDS address, tracked by ASYNCcnt.
// LDS byte address = low 32 bits of the generic pointer (flat LDS aperture
// truncates to the wave-relative LDS offset, ISA §10.2).
__device__ __forceinline__ void async_b128(const __bf16* gsrc, __bf16* ldst) {
  unsigned l = (unsigned)(size_t)(void*)ldst;
  asm volatile("global_load_async_to_lds_b128 %0, %1, off"
               :: "v"(l), "v"((unsigned long long)(size_t)(const void*)gsrc)
               : "memory");
}
__device__ __forceinline__ void wait_async() {
  asm volatile("s_wait_asynccnt 0x0" ::: "memory");
}
// width-32 tiles (GEMM K-step): NC chunks of 8 bf16 per thread, 128 threads
template <int NC>
__device__ __forceinline__ void acp_t32(__bf16* dst, const __bf16* src,
                                        size_t stride, int tid) {
#pragma unroll
  for (int i = 0; i < NC; ++i) {
    int j = i * 128 + tid;
    async_b128(src + (size_t)(j >> 2) * stride + (j & 3) * 8,
               dst + (j >> 2) * 32 + (j & 3) * 8);
  }
}
// 64x64 tiles (attention): 4 chunks per thread, 128 threads
__device__ __forceinline__ void acp_t64(__bf16* dst, const __bf16* src,
                                        size_t stride, int tid) {
#pragma unroll
  for (int i = 0; i < 4; ++i) {
    int j = i * 128 + tid;
    async_b128(src + (size_t)(j >> 3) * stride + (j & 7) * 8,
               dst + (j >> 3) * 64 + (j & 7) * 8);
  }
}

// ---------------------------------------------------------------------------
// Prep 1: f32 -> bf16, contiguous (for x). 8 elems/thread.
// ---------------------------------------------------------------------------
__global__ __launch_bounds__(256) void cvt_bf16_kernel(const float* __restrict__ in,
                                                       __bf16* __restrict__ out) {
  size_t i = ((size_t)blockIdx.x * 256 + threadIdx.x) * 8;
  float4 a = *(const float4*)(in + i);
  float4 b = *(const float4*)(in + i + 4);
  Pack8 pk;
  pk.h[0] = (__bf16)a.x; pk.h[1] = (__bf16)a.y;
  pk.h[2] = (__bf16)a.z; pk.h[3] = (__bf16)a.w;
  pk.h[4] = (__bf16)b.x; pk.h[5] = (__bf16)b.y;
  pk.h[6] = (__bf16)b.z; pk.h[7] = (__bf16)b.w;
  *(uint4*)(out + i) = pk.v;
}

// ---------------------------------------------------------------------------
// Prep 2: W [in][out] f32 -> WT [out][in] bf16, 64x64 LDS-tiled transpose.
// grid = (16, 16, 4); block = 256.
// ---------------------------------------------------------------------------
__global__ __launch_bounds__(256) void wtr_kernel(
    const float* __restrict__ Wq, const float* __restrict__ Wk,
    const float* __restrict__ Wv, const float* __restrict__ Wo,
    __bf16* __restrict__ Tq, __bf16* __restrict__ Tk,
    __bf16* __restrict__ Tv, __bf16* __restrict__ To) {
  __shared__ float sT[64][65];
  const int z = blockIdx.z;
  const float* W = (z == 0) ? Wq : (z == 1) ? Wk : (z == 2) ? Wv : Wo;
  __bf16*      T = (z == 0) ? Tq : (z == 1) ? Tk : (z == 2) ? Tv : To;
  const int k0 = blockIdx.x * 64, n0 = blockIdx.y * 64;
  const int t = threadIdx.x;
#pragma unroll
  for (int i = 0; i < 16; ++i) {       // coalesced f32 reads along n
    int e = i * 256 + t;
    sT[e >> 6][e & 63] = W[(size_t)(k0 + (e >> 6)) * D_MODEL + n0 + (e & 63)];
  }
  __syncthreads();
  const int n = t >> 2, kb = (t & 3) * 16;   // 16 contiguous k per thread
  Pack8 p0, p1;
#pragma unroll
  for (int j = 0; j < 8; ++j) {
    p0.h[j] = (__bf16)sT[kb + j][n];
    p1.h[j] = (__bf16)sT[kb + 8 + j][n];
  }
  uint4* dst = (uint4*)(T + (size_t)(n0 + n) * D_MODEL + k0 + kb);
  dst[0] = p0.v;
  dst[1] = p1.v;
}

// ---------------------------------------------------------------------------
// Kernel 1: QKV projection. grid = (M/128, D/64, 3); block = 128 (4 waves).
// Async double-buffered LDS; each wave computes 32x64 (8 WMMA/K-step).
// ---------------------------------------------------------------------------
__global__ __launch_bounds__(128) void qkv_proj_kernel(
    const __bf16* __restrict__ xb,
    const __bf16* __restrict__ Tq, const float* __restrict__ bq,
    const __bf16* __restrict__ Tk, const float* __restrict__ bk,
    const __bf16* __restrict__ Tv, const float* __restrict__ bv,
    __bf16* __restrict__ Qb, __bf16* __restrict__ Kb, __bf16* __restrict__ Vt) {
  __shared__ __bf16 sA[2][128 * 32];
  __shared__ __bf16 sB[2][64 * 32];

  const int tid = threadIdx.x, lane = tid & 31, wv = tid >> 5;
  const int m0 = blockIdx.x * 128, n0 = blockIdx.y * 64;
  const int z = blockIdx.z;
  const __bf16* T  = (z == 0) ? Tq : (z == 1) ? Tk : Tv;
  const float*  bs = (z == 0) ? bq : (z == 1) ? bk : bv;
  __bf16* out      = (z == 0) ? Qb : (z == 1) ? Kb : Vt;

  const __bf16* Abase = xb + (size_t)m0 * D_MODEL;
  const __bf16* Bbase = T + (size_t)n0 * D_MODEL;

  const v8f vzero = {0.f, 0.f, 0.f, 0.f, 0.f, 0.f, 0.f, 0.f};
  v8f acc[2][4];
#pragma unroll
  for (int i = 0; i < 2; ++i)
#pragma unroll
    for (int j = 0; j < 4; ++j) acc[i][j] = vzero;

  acp_t32<4>(sA[0], Abase, D_MODEL, tid);
  acp_t32<2>(sB[0], Bbase, D_MODEL, tid);
  wait_async();
  __syncthreads();

  int buf = 0;
  const int NK = D_MODEL / 32;
  for (int ks = 0; ks < NK; ++ks) {
    if (ks + 1 < NK) {   // async engine fills buf^1 while we compute on buf
      acp_t32<4>(sA[buf ^ 1], Abase + (ks + 1) * 32, D_MODEL, tid);
      acp_t32<2>(sB[buf ^ 1], Bbase + (ks + 1) * 32, D_MODEL, tid);
    }
    if (ks + 2 < NK) {   // prime caches -> global_prefetch_b8
      __builtin_prefetch(Abase + (size_t)(tid >> 2) * D_MODEL + (ks + 2) * 32 + (tid & 3) * 8, 0, 3);
      __builtin_prefetch(Bbase + (size_t)(tid >> 2) * D_MODEL + (ks + 2) * 32 + (tid & 3) * 8, 0, 3);
    }
    const __bf16* pA = sA[buf];
    const __bf16* pB = sB[buf];
    v16bf a0 = ld_a(pA + (wv * 32) * 32, 32, lane);
    v16bf a1 = ld_a(pA + (wv * 32 + 16) * 32, 32, lane);
    v16bf b0 = ld_b(pB + 0 * 16 * 32, 32, lane);
    v16bf b1 = ld_b(pB + 1 * 16 * 32, 32, lane);
    v16bf b2 = ld_b(pB + 2 * 16 * 32, 32, lane);
    v16bf b3 = ld_b(pB + 3 * 16 * 32, 32, lane);
    acc[0][0] = wmma_bf16(a0, b0, acc[0][0]);
    acc[0][1] = wmma_bf16(a0, b1, acc[0][1]);
    acc[0][2] = wmma_bf16(a0, b2, acc[0][2]);
    acc[0][3] = wmma_bf16(a0, b3, acc[0][3]);
    acc[1][0] = wmma_bf16(a1, b0, acc[1][0]);
    acc[1][1] = wmma_bf16(a1, b1, acc[1][1]);
    acc[1][2] = wmma_bf16(a1, b2, acc[1][2]);
    acc[1][3] = wmma_bf16(a1, b3, acc[1][3]);
    if (ks + 1 < NK) wait_async();
    __syncthreads();
    buf ^= 1;
  }

  const int lane15 = lane & 15, hf = lane >> 4;
#pragma unroll
  for (int ms = 0; ms < 2; ++ms) {
#pragma unroll
    for (int nt = 0; nt < 4; ++nt) {
      int n = n0 + nt * 16 + lane15;
      int h = n >> 6, dh = n & 63;
      float bias = bs[n];
#pragma unroll
      for (int r = 0; r < 8; ++r) {
        int m = m0 + wv * 32 + ms * 16 + r + 8 * hf;
        int bi = m / SEQ_L, l = m % SEQ_L;
        float val = acc[ms][nt][r] + bias;
        size_t idx = (z == 2)
            ? (((size_t)bi * NHEAD + h) * D_HEAD + dh) * SEQ_L + l   // V^T
            : (((size_t)bi * NHEAD + h) * SEQ_L + l) * D_HEAD + dh;  // Q/K
        out[idx] = (__bf16)val;
      }
    }
  }
}

// ---------------------------------------------------------------------------
// Kernel 2: flash attention. grid = (L/64, B*H); block = 128 (4 waves).
// Async double-buffered K/V tiles; each wave owns 16 query rows.
// ---------------------------------------------------------------------------
__global__ __launch_bounds__(128) void attn_kernel(
    const __bf16* __restrict__ Qb, const __bf16* __restrict__ Kb,
    const __bf16* __restrict__ Vt, __bf16* __restrict__ ctx) {
  __shared__ __bf16 sQ[64 * 64];         // [q][d]
  __shared__ __bf16 sK[2][64 * 64];      // [key][d]  == B^T for Q.K^T
  __shared__ __bf16 sV[2][64 * 64];      // [d][key]  == B^T for P.V
  __shared__ __bf16 sP[4][16 * 64];      // per-wave probs re-layout buffer

  const int tid = threadIdx.x, lane = tid & 31, wv = tid >> 5;
  const int lane15 = lane & 15, hf = lane >> 4;
  const int q0 = blockIdx.x * 64;
  const int bh = blockIdx.y;
  const int bi = bh >> 4, h = bh & 15;

  const __bf16* Qh = Qb + (size_t)bh * SEQ_L * D_HEAD;
  const __bf16* Kh = Kb + (size_t)bh * SEQ_L * D_HEAD;
  const __bf16* Vh = Vt + (size_t)bh * D_HEAD * SEQ_L;

  acp_t64(sQ, Qh + (size_t)q0 * D_HEAD, D_HEAD, tid);   // contiguous 8KB tile
  acp_t64(sK[0], Kh, D_HEAD, tid);
  acp_t64(sV[0], Vh, SEQ_L, tid);
  wait_async();
  __syncthreads();

  v16bf aq0 = ld_a(sQ + wv * 16 * 64, 64, lane);        // K = 0..31 of d
  v16bf aq1 = ld_a(sQ + wv * 16 * 64 + 32, 64, lane);   // K = 32..63 of d

  const v8f vzero = {0.f, 0.f, 0.f, 0.f, 0.f, 0.f, 0.f, 0.f};
  v8f o[4];
#pragma unroll
  for (int i = 0; i < 4; ++i) o[i] = vzero;
  float rmax[8], rsum[8];
#pragma unroll
  for (int r = 0; r < 8; ++r) { rmax[r] = -3.0e38f; rsum[r] = 0.f; }

  int buf = 0;
  const int NKB = SEQ_L / 64;
  for (int kb = 0; kb < NKB; ++kb) {
    if (kb + 1 < NKB) {
      acp_t64(sK[buf ^ 1], Kh + (size_t)(kb + 1) * 64 * D_HEAD, D_HEAD, tid);
      acp_t64(sV[buf ^ 1], Vh + (size_t)(kb + 1) * 64, SEQ_L, tid);
    }
    if (kb + 2 < NKB) {
      __builtin_prefetch(Kh + (size_t)(kb + 2) * 64 * D_HEAD + tid * 32, 0, 3);
      __builtin_prefetch(Vh + (size_t)(tid >> 3) * SEQ_L + (kb + 2) * 64 + (tid & 7) * 8, 0, 3);
    }

    // S = Q . K^T  (16 rows x 64 keys per wave)
    const __bf16* pK = sK[buf];
    v16bf kb0 = ld_b(pK + 0 * 16 * 64, 64, lane);
    v16bf kb1 = ld_b(pK + 1 * 16 * 64, 64, lane);
    v16bf kb2 = ld_b(pK + 2 * 16 * 64, 64, lane);
    v16bf kb3 = ld_b(pK + 3 * 16 * 64, 64, lane);
    v16bf kc0 = ld_b(pK + 0 * 16 * 64 + 32, 64, lane);
    v16bf kc1 = ld_b(pK + 1 * 16 * 64 + 32, 64, lane);
    v16bf kc2 = ld_b(pK + 2 * 16 * 64 + 32, 64, lane);
    v16bf kc3 = ld_b(pK + 3 * 16 * 64 + 32, 64, lane);
    v8f s[4];
    s[0] = wmma_bf16(aq0, kb0, vzero);
    s[1] = wmma_bf16(aq0, kb1, vzero);
    s[2] = wmma_bf16(aq0, kb2, vzero);
    s[3] = wmma_bf16(aq0, kb3, vzero);
    s[0] = wmma_bf16(aq1, kc0, s[0]);
    s[1] = wmma_bf16(aq1, kc1, s[1]);
    s[2] = wmma_bf16(aq1, kc2, s[2]);
    s[3] = wmma_bf16(aq1, kc3, s[3]);

    // Online softmax. C-frag: lane col n = lane15, VGPR r -> row r + 8*hf.
    __bf16* pw = sP[wv];
#pragma unroll
    for (int r = 0; r < 8; ++r) {
      float s0 = s[0][r] * ATT_SCALE, s1 = s[1][r] * ATT_SCALE;
      float s2 = s[2][r] * ATT_SCALE, s3 = s[3][r] * ATT_SCALE;
      float bm = fmaxf(fmaxf(s0, s1), fmaxf(s2, s3));
#pragma unroll
      for (int msk = 1; msk <= 8; msk <<= 1) bm = fmaxf(bm, __shfl_xor(bm, msk, 32));
      float nm = fmaxf(rmax[r], bm);
      float cr = __expf(rmax[r] - nm);
      rmax[r] = nm;
      float p0 = __expf(s0 - nm), p1 = __expf(s1 - nm);
      float p2 = __expf(s2 - nm), p3 = __expf(s3 - nm);
      int prow = (r + 8 * hf) * 64 + lane15;
      pw[prow +  0] = (__bf16)p0;
      pw[prow + 16] = (__bf16)p1;
      pw[prow + 32] = (__bf16)p2;
      pw[prow + 48] = (__bf16)p3;
      float bsum = (p0 + p1) + (p2 + p3);
#pragma unroll
      for (int msk = 1; msk <= 8; msk <<= 1) bsum += __shfl_xor(bsum, msk, 32);
      rsum[r] = rsum[r] * cr + bsum;
#pragma unroll
      for (int nt = 0; nt < 4; ++nt) o[nt][r] *= cr;
    }

    // O += P . V
    const __bf16* pV = sV[buf];
    v16bf pa0 = ld_a(pw, 64, lane);
    v16bf pa1 = ld_a(pw + 32, 64, lane);
    v16bf vb0 = ld_b(pV + 0 * 16 * 64, 64, lane);
    v16bf vb1 = ld_b(pV + 1 * 16 * 64, 64, lane);
    v16bf vb2 = ld_b(pV + 2 * 16 * 64, 64, lane);
    v16bf vb3 = ld_b(pV + 3 * 16 * 64, 64, lane);
    v16bf vc0 = ld_b(pV + 0 * 16 * 64 + 32, 64, lane);
    v16bf vc1 = ld_b(pV + 1 * 16 * 64 + 32, 64, lane);
    v16bf vc2 = ld_b(pV + 2 * 16 * 64 + 32, 64, lane);
    v16bf vc3 = ld_b(pV + 3 * 16 * 64 + 32, 64, lane);
    o[0] = wmma_bf16(pa0, vb0, o[0]);
    o[1] = wmma_bf16(pa0, vb1, o[1]);
    o[2] = wmma_bf16(pa0, vb2, o[2]);
    o[3] = wmma_bf16(pa0, vb3, o[3]);
    o[0] = wmma_bf16(pa1, vc0, o[0]);
    o[1] = wmma_bf16(pa1, vc1, o[1]);
    o[2] = wmma_bf16(pa1, vc2, o[2]);
    o[3] = wmma_bf16(pa1, vc3, o[3]);

    if (kb + 1 < NKB) wait_async();
    __syncthreads();
    buf ^= 1;
  }

  float inv[8];
#pragma unroll
  for (int r = 0; r < 8; ++r) inv[r] = 1.0f / rsum[r];
#pragma unroll
  for (int nt = 0; nt < 4; ++nt) {
    int col = h * D_HEAD + nt * 16 + lane15;
#pragma unroll
    for (int r = 0; r < 8; ++r) {
      int row = bi * SEQ_L + q0 + wv * 16 + r + 8 * hf;
      ctx[(size_t)row * D_MODEL + col] = (__bf16)(o[nt][r] * inv[r]);
    }
  }
}

// ---------------------------------------------------------------------------
// Kernel 3: output projection, f32 result. grid = (M/128, D/64); block = 128.
// ---------------------------------------------------------------------------
__global__ __launch_bounds__(128) void out_proj_kernel(
    const __bf16* __restrict__ ctx, const __bf16* __restrict__ To,
    const float* __restrict__ bo, float* __restrict__ out) {
  __shared__ __bf16 sA[2][128 * 32];
  __shared__ __bf16 sB[2][64 * 32];
  const int tid = threadIdx.x, lane = tid & 31, wv = tid >> 5;
  const int m0 = blockIdx.x * 128, n0 = blockIdx.y * 64;

  const __bf16* Abase = ctx + (size_t)m0 * D_MODEL;
  const __bf16* Bbase = To + (size_t)n0 * D_MODEL;

  const v8f vzero = {0.f, 0.f, 0.f, 0.f, 0.f, 0.f, 0.f, 0.f};
  v8f acc[2][4];
#pragma unroll
  for (int i = 0; i < 2; ++i)
#pragma unroll
    for (int j = 0; j < 4; ++j) acc[i][j] = vzero;

  acp_t32<4>(sA[0], Abase, D_MODEL, tid);
  acp_t32<2>(sB[0], Bbase, D_MODEL, tid);
  wait_async();
  __syncthreads();

  int buf = 0;
  const int NK = D_MODEL / 32;
  for (int ks = 0; ks < NK; ++ks) {
    if (ks + 1 < NK) {
      acp_t32<4>(sA[buf ^ 1], Abase + (ks + 1) * 32, D_MODEL, tid);
      acp_t32<2>(sB[buf ^ 1], Bbase + (ks + 1) * 32, D_MODEL, tid);
    }
    if (ks + 2 < NK) {
      __builtin_prefetch(Abase + (size_t)(tid >> 2) * D_MODEL + (ks + 2) * 32 + (tid & 3) * 8, 0, 3);
      __builtin_prefetch(Bbase + (size_t)(tid >> 2) * D_MODEL + (ks + 2) * 32 + (tid & 3) * 8, 0, 3);
    }
    const __bf16* pA = sA[buf];
    const __bf16* pB = sB[buf];
    v16bf a0 = ld_a(pA + (wv * 32) * 32, 32, lane);
    v16bf a1 = ld_a(pA + (wv * 32 + 16) * 32, 32, lane);
    v16bf b0 = ld_b(pB + 0 * 16 * 32, 32, lane);
    v16bf b1 = ld_b(pB + 1 * 16 * 32, 32, lane);
    v16bf b2 = ld_b(pB + 2 * 16 * 32, 32, lane);
    v16bf b3 = ld_b(pB + 3 * 16 * 32, 32, lane);
    acc[0][0] = wmma_bf16(a0, b0, acc[0][0]);
    acc[0][1] = wmma_bf16(a0, b1, acc[0][1]);
    acc[0][2] = wmma_bf16(a0, b2, acc[0][2]);
    acc[0][3] = wmma_bf16(a0, b3, acc[0][3]);
    acc[1][0] = wmma_bf16(a1, b0, acc[1][0]);
    acc[1][1] = wmma_bf16(a1, b1, acc[1][1]);
    acc[1][2] = wmma_bf16(a1, b2, acc[1][2]);
    acc[1][3] = wmma_bf16(a1, b3, acc[1][3]);
    if (ks + 1 < NK) wait_async();
    __syncthreads();
    buf ^= 1;
  }

  const int lane15 = lane & 15, hf = lane >> 4;
#pragma unroll
  for (int ms = 0; ms < 2; ++ms) {
#pragma unroll
    for (int nt = 0; nt < 4; ++nt) {
      int n = n0 + nt * 16 + lane15;
      float bias = bo[n];
#pragma unroll
      for (int r = 0; r < 8; ++r) {
        int m = m0 + wv * 32 + ms * 16 + r + 8 * hf;
        out[(size_t)m * D_MODEL + n] = acc[ms][nt][r] + bias;
      }
    }
  }
}

// ---------------------------------------------------------------------------
extern "C" void kernel_launch(void* const* d_in, const int* in_sizes, int n_in,
                              void* d_out, int out_size, void* d_ws, size_t ws_size,
                              hipStream_t stream) {
  const float* x  = (const float*)d_in[0];
  const float* Wq = (const float*)d_in[1];
  const float* bq = (const float*)d_in[2];
  const float* Wk = (const float*)d_in[3];
  const float* bk = (const float*)d_in[4];
  const float* Wv = (const float*)d_in[5];
  const float* bv = (const float*)d_in[6];
  const float* Wo = (const float*)d_in[7];
  const float* bo = (const float*)d_in[8];
  float* out = (float*)d_out;

  const int M = in_sizes[0] / D_MODEL;     // B*L = 4096
  const int B = M / SEQ_L;                 // 2
  const size_t nTok = (size_t)M * D_MODEL; // 4M elements
  const size_t nW = (size_t)D_MODEL * D_MODEL;

  __bf16* xb  = (__bf16*)d_ws;       // [M, D]        8MB
  __bf16* Tq  = xb + nTok;           // [D, D] W^T    2MB each
  __bf16* Tk  = Tq + nW;
  __bf16* Tv  = Tk + nW;
  __bf16* To  = Tv + nW;
  __bf16* Qb  = To + nW;             // [B,H,L,d]     8MB each
  __bf16* Kb  = Qb + nTok;
  __bf16* Vt  = Kb + nTok;           // [B,H,d,L]
  __bf16* ctx = Vt + nTok;           // [B*L, D]      (total 48MB)

  cvt_bf16_kernel<<<(unsigned)(nTok / 2048), 256, 0, stream>>>(x, xb);
  wtr_kernel<<<dim3(D_MODEL / 64, D_MODEL / 64, 4), 256, 0, stream>>>(
      Wq, Wk, Wv, Wo, Tq, Tk, Tv, To);

  qkv_proj_kernel<<<dim3(M / 128, D_MODEL / 64, 3), 128, 0, stream>>>(
      xb, Tq, bq, Tk, bk, Tv, bv, Qb, Kb, Vt);

  attn_kernel<<<dim3(SEQ_L / 64, B * NHEAD), 128, 0, stream>>>(Qb, Kb, Vt, ctx);

  out_proj_kernel<<<dim3(M / 128, D_MODEL / 64), 128, 0, stream>>>(ctx, To, bo, out);
}
